// FFT_Block_28209345200529
// MI455X (gfx1250) — compile-verified, matrix operand order checked by reading, not run on hardware
//
#include <hip/hip_runtime.h>

typedef __attribute__((ext_vector_type(16))) _Float16 v16h;
typedef __attribute__((ext_vector_type(8)))  float    v8f;
typedef __attribute__((ext_vector_type(4)))  float    f4;

#define BATCH 16
#define CCH   384
#define CHID  64
#define CF    193
#define CFP   208      // 13 tiles of 16
#define HW    16384

// ws layout (float offsets)
#define WS_Y0 0        // [16][384]
#define WS_H  6144     // [16][64]
#define WS_Y  7168     // [16][384]
#define WS_S1 13312    // [16][208]
#define WS_S2 16640    // [16][208]

// ---------------- Kernel 1: global average pool (bandwidth-bound) ----------------
__global__ void avgpool_k(const float* __restrict__ x, float* __restrict__ y0) {
  __shared__ float red[8];
  const int idx = blockIdx.x;                       // flat (b*C + c)
  const f4* p = reinterpret_cast<const f4*>(x) + (size_t)idx * (HW / 4);
  const int t = threadIdx.x;
  float s = 0.f;
#pragma unroll
  for (int i = 0; i < 16; ++i) {
    f4 v = __builtin_nontemporal_load(p + t + i * 256);
    s += v.x + v.y + v.z + v.w;
  }
#pragma unroll
  for (int off = 16; off > 0; off >>= 1) s += __shfl_xor(s, off, 32);
  if ((t & 31) == 0) red[t >> 5] = s;
  __syncthreads();
  if (t == 0) {
    float tot = 0.f;
#pragma unroll
    for (int w = 0; w < 8; ++w) tot += red[w];
    y0[idx] = tot * (1.0f / (float)HW);
  }
}

// ---------------- WMMA 16x16 tile GEMM: out = A[16xK] * W^T ----------------
// W element (n,k) at W[n*wn + k*wk]; nvalid = valid columns in this tile.
__device__ __forceinline__ v8f gemm16(const float* __restrict__ A, int K,
                                      const float* __restrict__ W,
                                      int nvalid, long wn, long wk) {
  const int lane = threadIdx.x & 31;
  const int mrow = lane & 15;                 // A row (M) and B column (N)
  const int khi  = (lane >> 4) & 1;           // upper-half lanes use +8 K offset
  v8f acc = {};
  for (int kk = 0; kk < K; kk += 32) {
    v16h a, b;
#pragma unroll
    for (int v = 0; v < 8; ++v) {
      // 16-bit A layout: VGPR v -> K base = (v>>2)*16 + (v&3)*2 + khi*8
      const int k0 = kk + ((v >> 2) << 4) + ((v & 3) << 1) + (khi << 3);
      a[2 * v + 0] = (_Float16)A[mrow * K + k0];
      a[2 * v + 1] = (_Float16)A[mrow * K + k0 + 1];
      float w0 = 0.f, w1 = 0.f;
      if (mrow < nvalid) {
        w0 = W[(long)mrow * wn + (long)k0 * wk];
        w1 = W[(long)mrow * wn + (long)(k0 + 1) * wk];
      }
      b[2 * v + 0] = (_Float16)w0;
      b[2 * v + 1] = (_Float16)w1;
    }
    acc = __builtin_amdgcn_wmma_f32_16x16x32_f16(false, a, false, b,
                                                 (short)0, acc, false, false);
  }
  return acc;
}

// act: 0 = none, 1 = relu, 2 = sigmoid
__device__ __forceinline__ void store_tile(v8f acc, float* __restrict__ out, int ldo,
                                           const float* __restrict__ bias,
                                           int n0, int nvalid, int act) {
  const int lane  = threadIdx.x & 31;
  const int n     = lane & 15;
  const int mbase = (lane >> 4) ? 8 : 0;      // C/D layout: lanes 16-31 -> M += 8
  if (n < nvalid) {
    const float bv = bias[n0 + n];
#pragma unroll
    for (int r = 0; r < 8; ++r) {
      float v = acc[r] + bv;
      if (act == 1) v = fmaxf(v, 0.f);
      else if (act == 2) v = 1.f / (1.f + __expf(-v));
      out[(mbase + r) * ldo + n0 + n] = v;
    }
  }
}

// ---------------- Kernel 2: h = relu(y0 @ W1c^T + b1)  [16x64] ----------------
__global__ void fc1_k(const float* __restrict__ W1, const float* __restrict__ b1,
                      float* __restrict__ ws) {
  const int wave = threadIdx.x >> 5;          // 4 waves, one N-tile each
  const int n0 = wave * 16;
  // W1[n,k,1,1] = W1[(n*384+k)*9 + 4]
  const float* Wbase = W1 + 4 + (long)n0 * 384 * 9;
  v8f acc = gemm16(ws + WS_Y0, CCH, Wbase, 16, 384 * 9, 9);
  store_tile(acc, ws + WS_H, CHID, b1, n0, 16, 1);
}

// ---------------- Kernel 3: y = sigmoid(h @ W2c^T + b2)  [16x384] ----------------
__global__ void fc2_k(const float* __restrict__ W2, const float* __restrict__ b2,
                      float* __restrict__ ws) {
  const int wave = threadIdx.x >> 5;          // 8 waves, 24 tiles
  for (int t = wave; t < 24; t += 8) {
    const int n0 = t * 16;
    // W2[n,k,1,1] = W2[(n*64+k)*9 + 4]
    const float* Wbase = W2 + 4 + (long)n0 * 64 * 9;
    v8f acc = gemm16(ws + WS_H, CHID, Wbase, 16, 64 * 9, 9);
    store_tile(acc, ws + WS_Y, CCH, b2, n0, 16, 2);
  }
}

// ---------------- Kernel 4: s1 = relu(y@Ws1^T+bs1), s2 = relu(y@Ws2^T+bs2) ----------------
__global__ void seq_k(const float* __restrict__ Ws1, const float* __restrict__ bs1,
                      const float* __restrict__ Ws2, const float* __restrict__ bs2,
                      float* __restrict__ ws) {
  const int wave = threadIdx.x >> 5;          // 8 waves, 26 tile-jobs
  for (int t = wave; t < 26; t += 8) {
    const int which = (t < 13) ? 0 : 1;
    const int tt = (t < 13) ? t : t - 13;
    const int n0 = tt * 16;
    const int nvalid = (CF - n0) < 16 ? (CF - n0) : 16;
    const float* W  = (which == 0) ? Ws1 : Ws2;     // [193,384] row-major
    const float* bb = (which == 0) ? bs1 : bs2;
    float* out = ws + ((which == 0) ? WS_S1 : WS_S2);
    v8f acc = gemm16(ws + WS_Y, CCH, W + (long)n0 * CCH, nvalid, CCH, 1);
    store_tile(acc, out, CFP, bb, n0, nvalid, 1);
  }
}

// ---------------- Kernel 5: rfft -> amp/phase modulation -> irfft -> out = xr*y ----------------
__global__ void fft_k(const float* __restrict__ ws, float* __restrict__ out) {
  __shared__ float ly[CCH];
  __shared__ float twc[CCH], tws[CCH];
  __shared__ float recR[CF], recI[CF];
  const int b = blockIdx.x;
  const int t = threadIdx.x;                  // 384 threads
  const float twopi_n = 6.283185307179586f / (float)CCH;
  ly[t] = ws[WS_Y + b * CCH + t];
  { float sn, cs; __sincosf(twopi_n * (float)t, &sn, &cs); tws[t] = sn; twc[t] = cs; }
  __syncthreads();

  if (t < CF) {
    // forward DFT bin t: yf[t] = sum_n y[n] * e^{-2pi i t n / 384}
    float re = 0.f, im = 0.f;
    int idx = 0;
    for (int n = 0; n < CCH; ++n) {
      re += ly[n] * twc[idx];
      im -= ly[n] * tws[idx];
      idx += t; if (idx >= CCH) idx -= CCH;
    }
    const float s1v = ws[WS_S1 + b * CFP + t];
    const float s2v = ws[WS_S2 + b * CFP + t];
    const float amp = sqrtf(re * re + im * im) * s1v;
    const float pha = atan2f(im, re) * s2v;
    float sp, cp; __sincosf(pha, &sp, &cp);
    recR[t] = amp * cp;
    recI[t] = amp * sp;
  }
  __syncthreads();

  // irfft at n = t (imag of bins 0 and 192 ignored, pocketfft C2R semantics)
  float acc = recR[0] + ((t & 1) ? -recR[CF - 1] : recR[CF - 1]);
  int idx = t % CCH;
  int cur = idx;
  for (int k = 1; k < CF - 1; ++k) {
    acc += 2.f * (recR[k] * twc[cur] - recI[k] * tws[cur]);
    cur += idx; if (cur >= CCH) cur -= CCH;
  }
  const float xr = acc * (1.f / (float)CCH);
  out[b * CCH + t] = xr * ly[t];
}

extern "C" void kernel_launch(void* const* d_in, const int* in_sizes, int n_in,
                              void* d_out, int out_size, void* d_ws, size_t ws_size,
                              hipStream_t stream) {
  const float* x   = (const float*)d_in[0];
  const float* W1  = (const float*)d_in[1];
  const float* b1  = (const float*)d_in[2];
  const float* W2  = (const float*)d_in[3];
  const float* b2  = (const float*)d_in[4];
  const float* Ws1 = (const float*)d_in[5];
  const float* bs1 = (const float*)d_in[6];
  const float* Ws2 = (const float*)d_in[7];
  const float* bs2 = (const float*)d_in[8];
  float* out = (float*)d_out;
  float* ws  = (float*)d_ws;

  avgpool_k<<<BATCH * CCH, 256, 0, stream>>>(x, ws + WS_Y0);
  fc1_k<<<1, 128, 0, stream>>>(W1, b1, ws);
  fc2_k<<<1, 256, 0, stream>>>(W2, b2, ws);
  seq_k<<<1, 256, 0, stream>>>(Ws1, bs1, Ws2, bs2, ws);
  fft_k<<<BATCH, CCH, 0, stream>>>(ws, out);
}
